// EpLSTM_Layer_5171140624764
// MI455X (gfx1250) — matrix operand order repaired and checked by reading, not compiled, not run on hardware
//
#include <hip/hip_runtime.h>
#include <hip/hip_bf16.h>

typedef __attribute__((ext_vector_type(16))) __bf16 v16bf;
typedef __attribute__((ext_vector_type(8)))  float  v8f;
typedef unsigned int       u32;
typedef unsigned short     u16;
typedef unsigned long long u64;

#define T_LEN  1024
#define BATCH  64
#define IN_DIM 256
#define HID    256
#define NWG    16        // persistent workgroups; WG w owns h-columns [16w,16w+16)
#define TPB    640       // 20 waves: wave = (gate g 0..4)*4 + (m-tile 0..3)

// ---- workspace layout (bytes) ----
#define FRAG_BYTES (80 * 8 * 512 * 2)        // per weight: 80 n-tiles x 8 k-tiles x 1KB
#define WIH_OFF    0
#define WHH_OFF    (FRAG_BYTES)
#define HBUF_OFF   (2 * FRAG_BYTES)          // bf16 h broadcast, 2 parity slabs [64][256]
#define HSLAB_ELEM (BATCH * HID)             // u16 elements per slab
#define HBUF_BYTES (2 * HSLAB_ELEM * 2)
#define FLAG_OFF   (HBUF_OFF + HBUF_BYTES)   // u32 flags[16*16] (one 64B line per WG)

__device__ __forceinline__ u16 f2bf(float f) {
  u32 u = __builtin_bit_cast(u32, f);
  u += 0x7FFFu + ((u >> 16) & 1u);           // round-to-nearest-even
  return (u16)(u >> 16);
}
__device__ __forceinline__ float sigf(float x)  { return 1.0f / (1.0f + __expf(-x)); }
__device__ __forceinline__ float tanha(float x) { float e = __expf(2.0f * x); return (e - 1.0f) / (e + 1.0f); }
__device__ __forceinline__ v16bf as16(uint4 lo, uint4 hi) {
  union { uint4 q[2]; v16bf v; } u; u.q[0] = lo; u.q[1] = hi; return u.v;
}

// ---------------- phase 0a: pack W_ih / W_hh (f32) into bf16 WMMA B-fragments ----------------
// B tile (K=32 x N=16): lane l holds column N=l&15; element e (0..15) holds K=(l<16?0:8)+e for e<8, K=16+(l<16?0:8)+(e-8) for e>=8.
__global__ void frag_build(const float* __restrict__ Wih, const float* __restrict__ Whh,
                           u16* __restrict__ wsb) {
  int i = blockIdx.x * blockDim.x + threadIdx.x;
  const int per = 80 * 8 * 32 * 16;
  if (i >= 2 * per) return;
  int a = (i >= per);
  int j = a ? (i - per) : i;
  const float* W = a ? Whh : Wih;            // [1280][256] row-major; B = W^T tile
  u16* F = wsb + (a ? (WHH_OFF / 2) : (WIH_OFF / 2));
  int e    = j & 15;
  int lane = (j >> 4) & 31;
  int kt   = (j >> 9) & 7;
  int nt   = j >> 12;
  int row  = nt * 16 + (lane & 15);
  int kk   = ((lane & 16) ? 8 : 0) + (e < 8 ? e : e + 8);
  F[j] = f2bf(W[row * 256 + kt * 32 + kk]);
}

// ---------------- phase 0b: init h slab 0 + flags ----------------
__global__ void init_state(const float* __restrict__ h0, u16* __restrict__ hbuf,
                           u32* __restrict__ flags) {
  int i = blockIdx.x * blockDim.x + threadIdx.x;
  if (i < BATCH * HID) hbuf[i] = f2bf(h0[i]);   // slab 0 (read at t=0)
  if (i < NWG * 16) flags[i] = 0u;
}

// ---- batched staging helpers (issue all loads, single wait, then LDS stores) ----
__device__ __forceinline__ void stage_x(const float* __restrict__ xt, int tid,
                                        u16 (*__restrict__ xs)[264]) {
  float4 xv[4][2];
#pragma unroll
  for (int u = 0; u < 4; ++u) {
    int i = tid + u * TPB;                   // chunk of 8 floats; 2048 chunks total
    if (u < 3 || i < 2048) {
      const float4* p = (const float4*)(xt + i * 8);
      xv[u][0] = p[0]; xv[u][1] = p[1];
    }
  }
#pragma unroll
  for (int u = 0; u < 4; ++u) {
    int i = tid + u * TPB;
    if (u < 3 || i < 2048) {
      int off = i * 8, row = off >> 8, col = off & 255;
      uint4 pk;
      pk.x = (u32)f2bf(xv[u][0].x) | ((u32)f2bf(xv[u][0].y) << 16);
      pk.y = (u32)f2bf(xv[u][0].z) | ((u32)f2bf(xv[u][0].w) << 16);
      pk.z = (u32)f2bf(xv[u][1].x) | ((u32)f2bf(xv[u][1].y) << 16);
      pk.w = (u32)f2bf(xv[u][1].z) | ((u32)f2bf(xv[u][1].w) << 16);
      *(uint4*)&xs[row][col] = pk;
    }
  }
}

__device__ __forceinline__ void stage_h(const u64* hslab, int tid,
                                        u16 (*__restrict__ hs)[264]) {
  u64 hv[7];                                  // 4096 u64 words over 640 threads
#pragma unroll
  for (int u = 0; u < 7; ++u) {
    int i = tid + u * TPB;
    if (u < 6 || i < 4096)
      hv[u] = __hip_atomic_load(&hslab[i], __ATOMIC_RELAXED, __HIP_MEMORY_SCOPE_AGENT);
  }
#pragma unroll
  for (int u = 0; u < 7; ++u) {
    int i = tid + u * TPB;
    if (u < 6 || i < 4096) {
      int row = i >> 6, c4 = i & 63;          // u64 = 4 bf16
      *(u64*)&hs[row][c4 * 4] = hv[u];
    }
  }
}

// ---------------- phase 2: persistent recurrent scan ----------------
__global__ __launch_bounds__(TPB, 1) void eplstm_scan(
    const float* __restrict__ x,    const float* __restrict__ mret,
    const float* __restrict__ c0,   const float* __restrict__ bvec,
    const u16*  __restrict__ Fih,   const u16*  __restrict__ Fhh,
    u16* __restrict__ hbuf, u32* __restrict__ flags, float* __restrict__ out) {
  __shared__ u16   xs[64][264];              // bf16 x_t staging (padded pitch)
  __shared__ u16   hs[64][264];              // bf16 h staging
  __shared__ float gs[20][16][17];           // per-wave 16x16 f32 gate tiles
  __shared__ u16   bihs[5][8][512];          // W_ih B-fragments, shared by 4 m-waves/gate

  const int tid  = threadIdx.x;
  const int wg   = blockIdx.x;
  const int lane = tid & 31;
  const int wv   = tid >> 5;
  const int mw   = wv & 3;
  const int gw   = wv >> 2;
  const int colbase = wg * 16;

  // ---- W_hh B-fragments live in VGPRs for all 1024 steps (64 VGPRs) ----
  v16bf Bhh[8];
  {
    const u16* phh = Fhh + (gw * 16 + wg) * 4096 + lane * 16;
#pragma unroll
    for (int k = 0; k < 8; ++k) Bhh[k] = *(const v16bf*)(phh + k * 512);
  }
  // ---- W_ih B-fragments into LDS (one copy per gate; 40KB) ----
  for (int ci = tid; ci < 2560; ci += TPB) { // 16B chunks: 5 gates x 4096 u16 / 8
    int g = ci >> 9, rem = ci & 511;
    *(uint4*)&bihs[g][0][rem * 8] =
        *(const uint4*)(Fih + (g * 16 + wg) * 4096 + rem * 8);
  }

  // ---- per-thread recurrent cell state (element<->thread map is step-invariant) ----
  const int nel = (tid < 384) ? 2 : 1;       // 1024 slice elements over 640 threads
  float creg[2]; float bia[2][5]; int er[2], ec[2];
#pragma unroll
  for (int q = 0; q < 2; ++q) {
    int idx = tid + q * TPB;
    if (idx < BATCH * 16) {
      int r = idx >> 4, cj = idx & 15;
      er[q] = r; ec[q] = cj;
      creg[q] = c0[r * HID + colbase + cj];
#pragma unroll
      for (int g = 0; g < 5; ++g) bia[q][g] = bvec[g * 256 + colbase + cj];
    }
  }

  const int rA = (mw << 4) + (lane & 15);
  const int cb = (lane & 16) ? 8 : 0;        // A layout: lanes16-31 start at K=8 / K=24
  const u16* pb = &bihs[gw][0][lane * 16];

  stage_x(x, tid, xs);                       // xs for t=0
  __syncthreads();                           // xs + bihs visible before first x-part

  for (int t = 0; t < T_LEN; ++t) {
    // issue m[t] loads early so latency hides under the x-part GEMM
    float mv[2];
#pragma unroll
    for (int q = 0; q < 2; ++q)
      if (q < nel)
        mv[q] = mret[((size_t)t * BATCH + er[q]) * HID + colbase + ec[q]];

    // ---- x-part GEMM first: independent of h, hides flag propagation + poll ----
    v8f acc = {0.f, 0.f, 0.f, 0.f, 0.f, 0.f, 0.f, 0.f};
    {
      uint4 xl = *(const uint4*)&xs[rA][cb];
      uint4 xh = *(const uint4*)&xs[rA][cb + 16];
      uint4 bl = *(const uint4*)(pb);
      uint4 bh = *(const uint4*)(pb + 8);
#pragma unroll
      for (int k = 0; k < 8; ++k) {
        uint4 nxl = xl, nxh = xh, nbl = bl, nbh = bh;
        if (k < 7) {
          int c1 = ((k + 1) << 5) + cb;
          nxl = *(const uint4*)&xs[rA][c1];
          nxh = *(const uint4*)&xs[rA][c1 + 16];
          nbl = *(const uint4*)(pb + (k + 1) * 512);
          nbh = *(const uint4*)(pb + (k + 1) * 512 + 8);
        }
        acc = __builtin_amdgcn_wmma_f32_16x16x32_bf16(false, as16(xl, xh), false, as16(bl, bh),
                                                      (short)0, acc, false, false);
        xl = nxl; xh = nxh; bl = nbl; bh = nbh;
      }
    }

    // ---- all-to-all flag barrier: 16 threads poll one WG flag each ----
    if (tid < NWG) {
      while (__hip_atomic_load(&flags[tid * 16], __ATOMIC_ACQUIRE, __HIP_MEMORY_SCOPE_AGENT)
             < (u32)t)
        __builtin_amdgcn_s_sleep(1);
    }
    __syncthreads();

    stage_h((const u64*)(hbuf + (t & 1) * HSLAB_ELEM), tid, hs);  // read parity slab
    __syncthreads();

    // ---- h-part GEMM: 8 WMMAs on the recurrent operand ----
    {
      uint4 hl = *(const uint4*)&hs[rA][cb];
      uint4 hh = *(const uint4*)&hs[rA][cb + 16];
#pragma unroll
      for (int k = 0; k < 8; ++k) {
        uint4 nhl = hl, nhh = hh;
        if (k < 7) {
          int c1 = ((k + 1) << 5) + cb;
          nhl = *(const uint4*)&hs[rA][c1];
          nhh = *(const uint4*)&hs[rA][c1 + 16];
        }
        acc = __builtin_amdgcn_wmma_f32_16x16x32_bf16(false, as16(hl, hh), false, Bhh[k],
                                                      (short)0, acc, false, false);
        hl = nhl; hh = nhh;
      }
    }
    {  // gate tile to LDS: C/D layout M = v + (lane>=16 ? 8 : 0), N = lane&15
      const int rb = (lane & 16) ? 8 : 0;
#pragma unroll
      for (int v = 0; v < 8; ++v) gs[wv][rb + v][lane & 15] = acc[v];
    }
    __syncthreads();

    // ---- elementwise cell update; write h into the OTHER parity slab ----
    u16* hw = hbuf + ((t + 1) & 1) * HSLAB_ELEM;
#pragma unroll
    for (int q = 0; q < 2; ++q) {
      if (q < nel) {
        int r = er[q], cj = ec[q];
        int lr = r & 15, mt = r >> 4;
        float gi = gs[0  + mt][lr][cj] + bia[q][0];
        float gf = gs[4  + mt][lr][cj] + bia[q][1];
        float gc = gs[8  + mt][lr][cj] + bia[q][2];
        float go = gs[12 + mt][lr][cj] + bia[q][3];
        float gr = gs[16 + mt][lr][cj] + bia[q][4];
        float c  = sigf(gf) * creg[q] + sigf(gi) * tanha(gc) + sigf(gr) * mv[q];
        float h  = sigf(go) * tanha(c);
        creg[q] = c;
        out[((size_t)t * BATCH + r) * HID + colbase + cj] = h;
        hw[r * HID + colbase + cj] = f2bf(h);
        if (t == T_LEN - 1) {
          size_t base = (size_t)T_LEN * BATCH * HID;
          out[base + r * HID + colbase + cj] = h;                 // h_T
          out[base + BATCH * HID + r * HID + colbase + cj] = c;   // c_T
        }
      }
    }

    // ---- stage x[t+1] (local; overlaps other WGs' progress) ----
    if (t + 1 < T_LEN) {
      stage_x(x + (size_t)(t + 1) * BATCH * IN_DIM, tid, xs);
      if (t + 2 < T_LEN)
        __builtin_prefetch(x + (size_t)(t + 2) * BATCH * IN_DIM, 0, 0);
    }

    // ---- publish h(t+1)-ready: release store to this WG's own flag line ----
    __threadfence();
    __syncthreads();                          // also orders xs writes before next x-part
    if (tid == 0)
      __hip_atomic_store(&flags[wg * 16], (u32)(t + 1), __ATOMIC_RELEASE,
                         __HIP_MEMORY_SCOPE_AGENT);
  }
}

extern "C" void kernel_launch(void* const* d_in, const int* in_sizes, int n_in,
                              void* d_out, int out_size, void* d_ws, size_t ws_size,
                              hipStream_t stream) {
  (void)in_sizes; (void)n_in; (void)out_size; (void)ws_size;
  const float* x   = (const float*)d_in[0];
  const float* mm  = (const float*)d_in[1];
  const float* h0  = (const float*)d_in[2];
  const float* c0  = (const float*)d_in[3];
  const float* Wih = (const float*)d_in[4];
  const float* Whh = (const float*)d_in[5];
  const float* b   = (const float*)d_in[6];
  float* out = (float*)d_out;
  char*  ws  = (char*)d_ws;
  u16* Fih   = (u16*)(ws + WIH_OFF);
  u16* Fhh   = (u16*)(ws + WHH_OFF);
  u16* hbuf  = (u16*)(ws + HBUF_OFF);
  u32* flags = (u32*)(ws + FLAG_OFF);

  frag_build<<<(2 * 327680 + 255) / 256, 256, 0, stream>>>(Wih, Whh, (u16*)ws);
  init_state<<<(BATCH * HID + 255) / 256, 256, 0, stream>>>(h0, hbuf, flags);
  eplstm_scan<<<NWG, TPB, 0, stream>>>(x, mm, c0, b, Fih, Fhh, hbuf, flags, out);
}
